// Block_15109694947394
// MI455X (gfx1250) — matrix operand order repaired and checked by reading, not compiled, not run on hardware
//
#include <hip/hip_runtime.h>
#include <hip/hip_bf16.h>
#include <math.h>

#define BB   4
#define TT   1024
#define DD   1024
#define HH   16
#define DHH  64
#define DFFN 4096
#define MM   (BB*TT)
#define QSZ  ((size_t)BB*HH*TT*DHH)

typedef __attribute__((ext_vector_type(16))) __bf16       v16bf;
typedef __attribute__((ext_vector_type(8)))  float        v8f;
typedef __attribute__((ext_vector_type(8)))  unsigned int u32x8;
typedef __attribute__((ext_vector_type(4)))  unsigned int u32x4;
typedef __attribute__((ext_vector_type(8)))  int          i32x8;
typedef __attribute__((ext_vector_type(4)))  int          i32x4;

// ---- toolchain feature detection for the Tensor Data Mover builtin -------------------
#if __has_include(<hip/amd_detail/amd_gfx1250_TDM.h>)
#define ATH_TDM_6ARG 1
#else
#define ATH_TDM_6ARG 0
#endif
#if defined(__has_builtin)
#if __has_builtin(__builtin_amdgcn_tensor_load_to_lds) && __has_builtin(__builtin_amdgcn_s_wait_tensorcnt)
#define ATH_HAS_TDM 1
#endif
#endif
#ifndef ATH_HAS_TDM
#define ATH_HAS_TDM 0
#endif

static __device__ __forceinline__ unsigned short f2b(float f) {
  unsigned int u = __builtin_bit_cast(unsigned int, f);
  unsigned int r = (u + 0x7FFFu + ((u >> 16) & 1u)) >> 16;
  return (unsigned short)r;
}
static __device__ __forceinline__ float sigm(float x) { return 1.0f / (1.0f + __expf(-x)); }

static __device__ __forceinline__ v8f vzero8() {
  v8f z;
#pragma unroll
  for (int i = 0; i < 8; ++i) z[i] = 0.0f;
  return z;
}

static __device__ __forceinline__ v8f wmma_bf16(v16bf a, v16bf b, v8f c) {
  return __builtin_amdgcn_wmma_f32_16x16x32_bf16(false, a, false, b, (short)0, c, false, false);
}

// A fragment: 16x32 bf16, M = lane&15. Per ISA 7.12.2 each lane's 16 elements are two
// contiguous 8-element runs: K = half*8 + [0..7] and K = 16 + half*8 + [0..7]
// -> two 16-byte vector loads.
static __device__ __forceinline__ v16bf load_matA(const unsigned short* A, int lda, int m0, int kk) {
  int lane = threadIdx.x & 31;
  int m = m0 + (lane & 15);
  int hlf = lane >> 4;
  const unsigned short* base = A + (size_t)m * lda + kk + hlf * 8;
  u32x4 lo = *(const u32x4*)(base);
  u32x4 hi = *(const u32x4*)(base + 16);
  u32x8 t;
#pragma unroll
  for (int i = 0; i < 4; ++i) { t[i] = lo[i]; t[4 + i] = hi[i]; }
  return __builtin_bit_cast(v16bf, t);
}

// B fragment: 32x16 bf16, N = lane&15, K = half*8 + [0..7] and 16 + half*8 + [0..7] -> W row-major [N,K]
static __device__ __forceinline__ v16bf load_matB(const unsigned short* W, int ldb, int n0, int kk) {
  int lane = threadIdx.x & 31;
  int n = n0 + (lane & 15);
  int khalf = lane >> 4;
  const unsigned short* base = W + (size_t)n * ldb + kk + khalf * 8;
  u32x4 lo = *(const u32x4*)(base);
  u32x4 hi = *(const u32x4*)(base + 16);
  u32x8 t;
#pragma unroll
  for (int i = 0; i < 4; ++i) { t[i] = lo[i]; t[4 + i] = hi[i]; }
  return __builtin_bit_cast(v16bf, t);
}

#if ATH_HAS_TDM
// TDM: DMA a 64(N) x 64(K) bf16 tile of row-major W[ld=K] into LDS (packed [64][64]).
// D# per CDNA5 ISA ch.8: group0 {count, lds_addr, global_addr, type=2},
// group1 {data_size=2B, tensor_dim0=K, tile 64x64, dim0_stride=K}.
static __device__ __forceinline__ void tdm_load_b_tile(unsigned lds_off,
                                                       const unsigned short* gptr, int K) {
  unsigned long long ga = (unsigned long long)(size_t)gptr;
  u32x4 g0;
  g0[0] = 1u;                                            // count=1, user descriptor
  g0[1] = lds_off;                                       // lds_addr (bytes)
  g0[2] = (unsigned)(ga & 0xFFFFFFFFull);                // global_addr[31:0]
  g0[3] = (unsigned)((ga >> 32) & 0x1FFFFFFull) | (2u << 30);  // addr[56:32], type=2
  unsigned td0 = (unsigned)K;                            // tensor_dim0 (row length, elems)
  unsigned td1 = 1u << 20;                               // generous row bound
  i32x8 g1;
  g1[0] = (int)(1u << 16);                               // data_size=1 -> 2 bytes
  g1[1] = (int)((td0 & 0xFFFFu) << 16);                  // tensor_dim0[15:0] @ bits63:48
  g1[2] = (int)(((td0 >> 16) & 0xFFFFu) | ((td1 & 0xFFFFu) << 16));
  g1[3] = (int)(((td1 >> 16) & 0xFFFFu) | (64u << 16));  // tile_dim0 = 64 (K)
  g1[4] = (int)64u;                                      // tile_dim1 = 64 (N), tile_dim2 = 0
  g1[5] = (int)(unsigned)K;                              // dim0_stride[31:0] (elems)
  g1[6] = 0;                                             // stride hi, dim1_stride lo
  g1[7] = 0;
  i32x4 z4 = {0, 0, 0, 0};
#if ATH_TDM_6ARG
  i32x8 z8 = {0, 0, 0, 0, 0, 0, 0, 0};
  __builtin_amdgcn_tensor_load_to_lds(g0, g1, z4, z4, z8, 0);
#else
  __builtin_amdgcn_tensor_load_to_lds(g0, g1, z4, z4, 0);
#endif
}
#endif

// ---------------- generic WMMA GEMM: C[M,N] = A[M,K] * W[N,K]^T (+bias)(+epilogue) ----
// B tile (64N x 64K) staged in LDS: TDM double-buffered DMA (wave 0 issues, TENSORcnt
// sync), or cooperative vector copy if the TDM builtin is unavailable.
// MODE 0: f32 out   MODE 2: GELU -> (f32 opt) + bf16 out   MODE 3: QKV scatter (V transposed)
// MODE 4: f32 out = resid + acc + bias
template <int MODE>
__global__ __launch_bounds__(256) void k_gemm(const unsigned short* __restrict__ A,
                                              const unsigned short* __restrict__ W,
                                              const float* __restrict__ bias,
                                              const float* __restrict__ resid,
                                              float* __restrict__ Cf,
                                              unsigned short* __restrict__ Cb,
                                              int N, int K) {
  __shared__ __attribute__((aligned(64))) unsigned short bstage[2][64 * 64];
  int wave = threadIdx.x >> 5;
  int m0 = blockIdx.y * 128 + wave * 16;
  int n0 = blockIdx.x * 64;
  v8f acc[4];
#pragma unroll
  for (int t = 0; t < 4; ++t) acc[t] = vzero8();
  const int steps = K >> 6;                              // K stage = 64

#if ATH_HAS_TDM
  unsigned ldsbase = (unsigned)(unsigned long long)(size_t)&bstage[0][0];
  if (threadIdx.x == 0) tdm_load_b_tile(ldsbase, W + (size_t)n0 * K, K);
  for (int s = 0; s < steps; ++s) {
    int kk = s << 6;
    if (threadIdx.x == 0) {
      if (s + 1 < steps) {
        tdm_load_b_tile(ldsbase + 8192u * (unsigned)((s + 1) & 1),
                        W + (size_t)n0 * K + kk + 64, K);
        __builtin_amdgcn_s_wait_tensorcnt(1);   // current tile done, next in flight
      } else {
        __builtin_amdgcn_s_wait_tensorcnt(0);
      }
    }
    __syncthreads();
    if (s + 1 < steps)
      __builtin_prefetch(A + (size_t)(m0 + (threadIdx.x & 15)) * K + kk + 64, 0, 0);
    const unsigned short* bt = &bstage[s & 1][0];
#pragma unroll
    for (int ks = 0; ks < 64; ks += 32) {
      v16bf a = load_matA(A, K, m0, kk + ks);
#pragma unroll
      for (int t = 0; t < 4; ++t)
        acc[t] = wmma_bf16(a, load_matB(bt, 64, t * 16, ks), acc[t]);
    }
    __syncthreads();
  }
#else
  for (int s = 0; s < steps; ++s) {
    int kk = s << 6;
    __syncthreads();
    {
      int r = threadIdx.x >> 2, c = (threadIdx.x & 3) * 16;
      *(u32x4*)&bstage[0][r * 64 + c] = *(const u32x4*)(W + (size_t)(n0 + r) * K + kk + c);
      *(u32x4*)&bstage[0][r * 64 + c + 8] = *(const u32x4*)(W + (size_t)(n0 + r) * K + kk + c + 8);
    }
    __syncthreads();
#pragma unroll
    for (int ks = 0; ks < 64; ks += 32) {
      v16bf a = load_matA(A, K, m0, kk + ks);
#pragma unroll
      for (int t = 0; t < 4; ++t)
        acc[t] = wmma_bf16(a, load_matB(&bstage[0][0], 64, t * 16, ks), acc[t]);
    }
  }
#endif

  int lane = threadIdx.x & 31, hlf = lane >> 4, nn = lane & 15;
#pragma unroll
  for (int t = 0; t < 4; ++t) {
#pragma unroll
    for (int r = 0; r < 8; ++r) {
      float v = acc[t][r];
      int row = m0 + r + 8 * hlf;
      int col = n0 + t * 16 + nn;
      if (bias) v += bias[col];
      if (MODE == 0) {
        Cf[(size_t)row * N + col] = v;
      } else if (MODE == 2) {
        float gl = 0.5f * v * (1.0f + erff(v * 0.70710678118654752f));
        if (Cf) Cf[(size_t)row * N + col] = gl;
        Cb[(size_t)row * N + col] = f2b(gl);
      } else if (MODE == 3) {
        int b = row >> 10, tt = row & (TT - 1);
        int sel = col >> 10;
        int h = (col >> 6) & (HH - 1);
        int d = col & (DHH - 1);
        size_t idx;
        if (sel < 2)
          idx = (size_t)sel * QSZ + ((size_t)(b * HH + h) * TT + tt) * DHH + d;   // q,k [B,H,T,DH]
        else
          idx = 2ull * QSZ + ((size_t)(b * HH + h) * DHH + d) * TT + tt;          // vT [B,H,DH,T]
        Cb[idx] = f2b(v);
      } else if (MODE == 4) {
        Cf[(size_t)row * N + col] = resid[(size_t)row * N + col] + v;
      }
    }
  }
}

// ---------------- LayerNorm -> f32 (opt) + bf16 --------------------------------------
__global__ __launch_bounds__(256) void k_layernorm(const float* __restrict__ x,
                                                   const float* __restrict__ g,
                                                   const float* __restrict__ be,
                                                   float* __restrict__ hf,
                                                   unsigned short* __restrict__ hb) {
  __shared__ float red[256];
  int row = blockIdx.x, tid = threadIdx.x;
  const float* xr = x + (size_t)row * DD;
  float v[4];
  float s = 0.0f;
#pragma unroll
  for (int i = 0; i < 4; ++i) { v[i] = xr[tid + 256 * i]; s += v[i]; }
  red[tid] = s; __syncthreads();
  for (int o = 128; o > 0; o >>= 1) { if (tid < o) red[tid] += red[tid + o]; __syncthreads(); }
  float mu = red[0] * (1.0f / DD);
  __syncthreads();
  s = 0.0f;
#pragma unroll
  for (int i = 0; i < 4; ++i) { float d = v[i] - mu; s += d * d; }
  red[tid] = s; __syncthreads();
  for (int o = 128; o > 0; o >>= 1) { if (tid < o) red[tid] += red[tid + o]; __syncthreads(); }
  float rstd = rsqrtf(red[0] * (1.0f / DD) + 1e-5f);
#pragma unroll
  for (int i = 0; i < 4; ++i) {
    int c = tid + 256 * i;
    float y = (v[i] - mu) * rstd * g[c] + be[c];
    if (hf) hf[(size_t)row * DD + c] = y;
    hb[(size_t)row * DD + c] = f2b(y);
  }
}

// ---------------- flash attention, 16-row Q tile per wave -----------------------------
__global__ __launch_bounds__(128) void k_attn(const unsigned short* __restrict__ qkv,
                                              float* __restrict__ seq_out) {
  __shared__ __attribute__((aligned(16))) unsigned short pshm[4][16 * 32];
  int wave = threadIdx.x >> 5;
  int lane = threadIdx.x & 31, hlf = lane >> 4, nn = lane & 15;
  int tile = blockIdx.x * 4 + wave;                 // 0 .. B*H*T/16-1
  int i0 = (tile & (TT / 16 - 1)) * 16;
  int bh = tile / (TT / 16);
  const unsigned short* qb = qkv + (size_t)bh * TT * DHH;
  const unsigned short* kb = qkv + QSZ + (size_t)bh * TT * DHH;
  const unsigned short* vtb = qkv + 2ull * QSZ + (size_t)bh * DHH * TT;

  v16bf aq0 = load_matA(qb, DHH, i0, 0);
  v16bf aq1 = load_matA(qb, DHH, i0, 32);

  float mrow[8], lrow[8];
  v8f o[4];
#pragma unroll
  for (int r = 0; r < 8; ++r) { mrow[r] = -1e30f; lrow[r] = 0.0f; }
#pragma unroll
  for (int t = 0; t < 4; ++t) o[t] = vzero8();

  for (int j0 = 0; j0 <= i0 + 15; j0 += 32) {       // wave-uniform causal bound
    v8f s0 = vzero8(), s1 = vzero8();
    s0 = wmma_bf16(aq0, load_matB(kb, DHH, j0, 0), s0);
    s0 = wmma_bf16(aq1, load_matB(kb, DHH, j0, 32), s0);
    s1 = wmma_bf16(aq0, load_matB(kb, DHH, j0 + 16, 0), s1);
    s1 = wmma_bf16(aq1, load_matB(kb, DHH, j0 + 16, 32), s1);
#pragma unroll
    for (int r = 0; r < 8; ++r) {
      int row = i0 + r + 8 * hlf;
      float x0 = (j0 + nn <= row) ? s0[r] * 0.125f : -1e30f;
      float x1 = (j0 + 16 + nn <= row) ? s1[r] * 0.125f : -1e30f;
      float mx = fmaxf(x0, x1);
#pragma unroll
      for (int off = 1; off < 16; off <<= 1) mx = fmaxf(mx, __shfl_xor(mx, off, 16));
      float mnew = fmaxf(mrow[r], mx);
      float p0 = __expf(x0 - mnew), p1 = __expf(x1 - mnew);
      float ps = p0 + p1;
#pragma unroll
      for (int off = 1; off < 16; off <<= 1) ps += __shfl_xor(ps, off, 16);
      float alpha = __expf(mrow[r] - mnew);
      lrow[r] = lrow[r] * alpha + ps;
      mrow[r] = mnew;
#pragma unroll
      for (int t = 0; t < 4; ++t) o[t][r] *= alpha;
      int row0 = r + 8 * hlf;
      pshm[wave][row0 * 32 + nn] = f2b(p0);
      pshm[wave][row0 * 32 + 16 + nn] = f2b(p1);
    }
    v16bf ap = load_matA(&pshm[wave][0], 32, 0, 0); // C-layout -> A-fragment via LDS
#pragma unroll
    for (int t = 0; t < 4; ++t)
      o[t] = wmma_bf16(ap, load_matB(vtb, TT, t * 16, j0), o[t]);
  }

  int b = bh >> 4, h = bh & 15;
#pragma unroll
  for (int t = 0; t < 4; ++t)
#pragma unroll
    for (int r = 0; r < 8; ++r) {
      int row = i0 + r + 8 * hlf;
      seq_out[((size_t)(b * TT + row)) * DD + h * DHH + t * 16 + nn] = o[t][r] / lrow[r];
    }
}

// ---------------- small elementwise / VALU kernels ------------------------------------
__global__ void k_cvt(const float* __restrict__ s, unsigned short* __restrict__ d, int n) {
  int i = blockIdx.x * 256 + threadIdx.x;
  if (i < n) d[i] = f2b(s[i]);
}

__global__ void k_shift(const unsigned short* __restrict__ hb, unsigned short* __restrict__ hp) {
  int i = blockIdx.x * 256 + threadIdx.x;          // over B*T*D
  int row = i >> 10;
  int t = row & (TT - 1);
  hp[i] = (t == 0) ? (unsigned short)0 : hb[i - DD];
}

// Plucker lines + J-dual; i indexes (b*T+t)*H + h
__global__ void k_lines(const float* __restrict__ w1, const float* __restrict__ w2,
                        const float* __restrict__ r1, const float* __restrict__ r2,
                        float* __restrict__ Jw, float* __restrict__ rdl) {
  int i = blockIdx.x * 256 + threadIdx.x;
  if (i >= MM * HH) return;
  const int pi[6] = {0, 0, 0, 1, 1, 2};
  const int pj[6] = {1, 2, 3, 2, 3, 3};
  {
    const float* a = w1 + (size_t)i * 4;
    const float* b = w2 + (size_t)i * 4;
    float L[6], s = 0.0f;
#pragma unroll
    for (int c = 0; c < 6; ++c) { L[c] = a[pi[c]] * b[pj[c]] - a[pj[c]] * b[pi[c]]; s += L[c] * L[c]; }
    float inv = 1.0f / fmaxf(sqrtf(s), 1e-12f);
    float* o = Jw + (size_t)i * 6;                  // J-dual: [L5,-L4,L3,L2,-L1,L0]
    o[0] = L[5] * inv; o[1] = -L[4] * inv; o[2] = L[3] * inv;
    o[3] = L[2] * inv; o[4] = -L[1] * inv; o[5] = L[0] * inv;
  }
  {
    const float* a = r1 + (size_t)i * 4;
    const float* b = r2 + (size_t)i * 4;
    float L[6], s = 0.0f;
#pragma unroll
    for (int c = 0; c < 6; ++c) { L[c] = a[pi[c]] * b[pj[c]] - a[pj[c]] * b[pi[c]]; s += L[c] * L[c]; }
    float inv = 1.0f / fmaxf(sqrtf(s), 1e-12f);
    float* o = rdl + (size_t)i * 6;
#pragma unroll
    for (int c = 0; c < 6; ++c) o[c] = L[c] * inv;
  }
}

// mem_score[b,t,h] = sum_{j<i} decay^(i-j) * |<read_i, Jw_j>|
__global__ void k_memscore(const float* __restrict__ rdl, const float* __restrict__ Jw,
                           const float* __restrict__ dlog, float* __restrict__ ms) {
  int idx = blockIdx.x * 256 + threadIdx.x;        // (b*T+i)*H + h
  if (idx >= MM * HH) return;
  int h = idx & (HH - 1);
  int row = idx >> 4;
  int b = row >> 10, i = row & (TT - 1);
  const float* rp = rdl + (size_t)idx * 6;
  float r0 = rp[0], r1 = rp[1], r2 = rp[2], r3 = rp[3], r4 = rp[4], r5 = rp[5];
  float dec = sigm(dlog[h]);
  float w = dec, s = 0.0f;
  const float* base = Jw + (((size_t)b * TT) * HH + h) * 6;
  for (int j = i - 1; j >= 0; --j) {
    const float* q = base + (size_t)j * HH * 6;
    float d = r0 * q[0] + r1 * q[1] + r2 * q[2] + r3 * q[3] + r4 * q[4] + r5 * q[5];
    s += w * fabsf(d);
    w *= dec;
  }
  ms[idx] = s;
}

__global__ void k_gate(const float* __restrict__ hf, const float* __restrict__ gw,
                       const float* __restrict__ gb, float* __restrict__ gate) {
  int idx = blockIdx.x * 256 + threadIdx.x;        // (b*T+t)*H + h
  if (idx >= MM * HH) return;
  int h = idx & (HH - 1);
  int row = idx >> 4;
  const float* xr = hf + (size_t)row * DD;
  const float* wr = gw + (size_t)h * DD;
  float s = gb[h];
  for (int c = 0; c < DD; c += 4)
    s += xr[c] * wr[c] + xr[c + 1] * wr[c + 1] + xr[c + 2] * wr[c + 2] + xr[c + 3] * wr[c + 3];
  gate[idx] = sigm(s);
}

__global__ void k_gated(const float* __restrict__ ms, const float* __restrict__ gate,
                        const float* __restrict__ scale, float* __restrict__ gated) {
  int row = blockIdx.x * 256 + threadIdx.x;
  if (row >= MM) return;
  float s = 0.0f;
#pragma unroll
  for (int h = 0; h < HH; ++h)
    s += sigm(ms[row * HH + h] * scale[h]) * gate[row * HH + h];
  gated[row] = s * (1.0f / HH);
}

__global__ void k_mix(const float* __restrict__ seq, const float* __restrict__ mv,
                      const float* __restrict__ gated, unsigned short* __restrict__ cb) {
  int i = blockIdx.x * 256 + threadIdx.x;          // B*T*D
  cb[i] = f2b(seq[i] + gated[i >> 10] * mv[i]);
}

// ---------------- host ----------------------------------------------------------------
extern "C" void kernel_launch(void* const* d_in, const int* in_sizes, int n_in,
                              void* d_out, int out_size, void* d_ws, size_t ws_size,
                              hipStream_t stream) {
  const float* x = (const float*)d_in[0];
  const float* ln1_g = (const float*)d_in[1];
  const float* ln1_b = (const float*)d_in[2];
  const float* qkv_w = (const float*)d_in[3];
  const float* qkv_b = (const float*)d_in[4];
  const float* w1w = (const float*)d_in[5];
  const float* w2w = (const float*)d_in[6];
  const float* w1r = (const float*)d_in[7];
  const float* w2r = (const float*)d_in[8];
  const float* memv_w = (const float*)d_in[9];
  const float* memv_b = (const float*)d_in[10];
  const float* memg_w = (const float*)d_in[11];
  const float* memg_b = (const float*)d_in[12];
  const float* mem_scale = (const float*)d_in[13];
  const float* out_w = (const float*)d_in[14];
  const float* out_b = (const float*)d_in[15];
  const float* decay_logits = (const float*)d_in[16];
  const float* ln2_g = (const float*)d_in[17];
  const float* ln2_b = (const float*)d_in[18];
  const float* fc1_w = (const float*)d_in[19];
  const float* fc1_b = (const float*)d_in[20];
  const float* fc2_w = (const float*)d_in[21];
  const float* fc2_b = (const float*)d_in[22];
  float* out = (float*)d_out;

  char* ws = (char*)d_ws;
  size_t off = 0;
  auto alloc = [&](size_t bytes) -> void* {
    off = (off + 255) & ~(size_t)255;
    void* p = ws + off;
    off += bytes;
    return p;
  };

  // bf16 weights
  unsigned short* wqkvb = (unsigned short*)alloc((size_t)3 * DD * DD * 2);
  unsigned short* w1b = (unsigned short*)alloc((size_t)64 * DD * 2);
  unsigned short* w2b = (unsigned short*)alloc((size_t)64 * DD * 2);
  unsigned short* r1b = (unsigned short*)alloc((size_t)64 * DD * 2);
  unsigned short* r2b = (unsigned short*)alloc((size_t)64 * DD * 2);
  unsigned short* wmvb = (unsigned short*)alloc((size_t)DD * DD * 2);
  unsigned short* woutb = (unsigned short*)alloc((size_t)DD * DD * 2);
  unsigned short* wfc1b = (unsigned short*)alloc((size_t)DFFN * DD * 2);
  unsigned short* wfc2b = (unsigned short*)alloc((size_t)DD * DFFN * 2);
  // activations
  float* hf = (float*)alloc((size_t)MM * DD * 4);
  unsigned short* hbf = (unsigned short*)alloc((size_t)MM * DD * 2);
  unsigned short* hpb = (unsigned short*)alloc((size_t)MM * DD * 2);
  unsigned short* qkvb = (unsigned short*)alloc(3 * QSZ * 2);
  float* w1f = (float*)alloc((size_t)MM * 64 * 4);
  float* w2f = (float*)alloc((size_t)MM * 64 * 4);
  float* r1f = (float*)alloc((size_t)MM * 64 * 4);
  float* r2f = (float*)alloc((size_t)MM * 64 * 4);
  float* Jw = (float*)alloc((size_t)MM * HH * 6 * 4);
  float* rdl = (float*)alloc((size_t)MM * HH * 6 * 4);
  float* ms = (float*)alloc((size_t)MM * HH * 4);
  float* gate = (float*)alloc((size_t)MM * HH * 4);
  float* gated = (float*)alloc((size_t)MM * 4);
  float* seq = (float*)alloc((size_t)MM * DD * 4);
  float* mv = (float*)alloc((size_t)MM * DD * 4);
  unsigned short* comb = (unsigned short*)alloc((size_t)MM * DD * 2);
  float* x2 = (float*)alloc((size_t)MM * DD * 4);
  unsigned short* h2b = (unsigned short*)alloc((size_t)MM * DD * 2);
  unsigned short* actb = (unsigned short*)alloc((size_t)MM * DFFN * 2);

  auto cvt = [&](const float* s, unsigned short* d, int n) {
    k_cvt<<<(n + 255) / 256, 256, 0, stream>>>(s, d, n);
  };
  cvt(qkv_w, wqkvb, 3 * DD * DD);
  cvt(w1w, w1b, 64 * DD);
  cvt(w2w, w2b, 64 * DD);
  cvt(w1r, r1b, 64 * DD);
  cvt(w2r, r2b, 64 * DD);
  cvt(memv_w, wmvb, DD * DD);
  cvt(out_w, woutb, DD * DD);
  cvt(fc1_w, wfc1b, DFFN * DD);
  cvt(fc2_w, wfc2b, DD * DFFN);

  // LN1 -> h (f32 + bf16), shifted copy for x_prev
  k_layernorm<<<MM, 256, 0, stream>>>(x, ln1_g, ln1_b, hf, hbf);
  k_shift<<<(MM * DD) / 256, 256, 0, stream>>>(hbf, hpb);

  // QKV GEMM with q/k/vT scatter
  k_gemm<3><<<dim3(3 * DD / 64, MM / 128), 256, 0, stream>>>(hbf, wqkvb, qkv_b, nullptr,
                                                             nullptr, qkvb, 3 * DD, DD);
  // small geometric projections (N=64)
  k_gemm<0><<<dim3(1, MM / 128), 256, 0, stream>>>(hpb, w1b, nullptr, nullptr, w1f, nullptr, 64, DD);
  k_gemm<0><<<dim3(1, MM / 128), 256, 0, stream>>>(hbf, w2b, nullptr, nullptr, w2f, nullptr, 64, DD);
  k_gemm<0><<<dim3(1, MM / 128), 256, 0, stream>>>(hbf, r1b, nullptr, nullptr, r1f, nullptr, 64, DD);
  k_gemm<0><<<dim3(1, MM / 128), 256, 0, stream>>>(hbf, r2b, nullptr, nullptr, r2f, nullptr, 64, DD);

  k_lines<<<(MM * HH) / 256, 256, 0, stream>>>(w1f, w2f, r1f, r2f, Jw, rdl);
  k_memscore<<<(MM * HH) / 256, 256, 0, stream>>>(rdl, Jw, decay_logits, ms);

  // flash attention (B*H*T/16 tiles, 4 waves/block)
  k_attn<<<(BB * HH * (TT / 16)) / 4, 128, 0, stream>>>(qkvb, seq);

  // memory value + gates + mix
  k_gemm<0><<<dim3(DD / 64, MM / 128), 256, 0, stream>>>(hbf, wmvb, memv_b, nullptr, mv, nullptr, DD, DD);
  k_gate<<<(MM * HH) / 256, 256, 0, stream>>>(hf, memg_w, memg_b, gate);
  k_gated<<<(MM + 255) / 256, 256, 0, stream>>>(ms, gate, mem_scale, gated);
  k_mix<<<(MM * DD) / 256, 256, 0, stream>>>(seq, mv, gated, comb);

  // out projection + residual -> x2
  k_gemm<4><<<dim3(DD / 64, MM / 128), 256, 0, stream>>>(comb, woutb, out_b, x, x2, nullptr, DD, DD);

  // LN2 -> bf16, MLP
  k_layernorm<<<MM, 256, 0, stream>>>(x2, ln2_g, ln2_b, nullptr, h2b);
  k_gemm<2><<<dim3(DFFN / 64, MM / 128), 256, 0, stream>>>(h2b, wfc1b, fc1_b, nullptr,
                                                           nullptr, actb, DFFN, DD);
  k_gemm<4><<<dim3(DD / 64, MM / 128), 256, 0, stream>>>(actb, wfc2b, fc2_b, x2, out, nullptr, DD, DFFN);

  (void)in_sizes; (void)n_in; (void)out_size; (void)ws_size;
}
